// AttLayer_14620068675632
// MI455X (gfx1250) — matrix operand order repaired and checked by reading, not compiled, not run on hardware
//
#include <hip/hip_runtime.h>
#include <hip/hip_bf16.h>

// Shapes fixed by the reference.
#define DD   256
#define TT   2048
#define BBATCH 64
#define LDSK 264   // padded k-stride (in bf16 elems) for transposed W in LDS

typedef __attribute__((ext_vector_type(16))) __bf16 bf16x16;
typedef __attribute__((ext_vector_type(8)))  __bf16 bf16x8;
typedef __attribute__((ext_vector_type(8)))  float  f32x8;

// ---------------------------------------------------------------------------
// Kernel 1: scores[b,t] = uw . tanh(x[b,t,:] @ W + b)
// One workgroup = 8 waves; each wave computes a 16-row tile of the GEMM with
// v_wmma_f32_16x16x32_bf16, fusing the tanh + uw-dot epilogue in registers.
// W is staged once per workgroup into LDS as bf16, transposed (Wt[e][d]) so
// each WMMA B-tile is two contiguous 16-byte LDS loads per lane.
// ---------------------------------------------------------------------------
__global__ __launch_bounds__(256) void attn_scores_kernel(
    const float* __restrict__ x, const float* __restrict__ W,
    const float* __restrict__ bvec, const float* __restrict__ uw,
    float* __restrict__ scores)
{
    extern __shared__ char smem_raw[];
    __bf16* Wt     = (__bf16*)smem_raw;                       // [DD][LDSK] bf16
    float*  b_lds  = (float*)(smem_raw + (size_t)DD * LDSK * 2);
    float*  uw_lds = b_lds + DD;

    const int tid = threadIdx.x;

    // Stage W transposed+converted: Wt[e*LDSK + d] = (bf16)W[d*DD + e].
    // Global reads coalesced along e (= tid); one-time strided LDS stores.
    for (int d = 0; d < DD; ++d) {
        Wt[tid * LDSK + d] = (__bf16)W[d * DD + tid];
    }
    b_lds[tid]  = bvec[tid];
    uw_lds[tid] = uw[tid];
    __syncthreads();

    const int lane = tid & 31;
    const int wave = tid >> 5;        // 0..7
    const int col  = lane & 15;       // N-index within a 16x16 C tile
    const int half = lane >> 4;       // 0 or 1
    const long rowBase = ((long)blockIdx.x * 8 + wave) * 16;
    const float* __restrict__ xrow = x + (rowBase + col) * DD;  // A row per lane

    f32x8 acc[16];
    #pragma unroll
    for (int n = 0; n < 16; ++n) acc[n] = (f32x8)0.0f;

    #pragma unroll
    for (int ks = 0; ks < 8; ++ks) {
        // ---- A tile (16x32 bf16): ISA layout -> lane half selects k+8 group.
        const int kbaseA = ks * 32 + half * 8;
        // Locality 3 -> WGP-scope prefetch (pull into all cache levels):
        // this wave consumes the next k-step's data itself.
        if (ks < 7) __builtin_prefetch(xrow + kbaseA + 32, 0, 3);  // global_prefetch_b8
        bf16x16 a;
        #pragma unroll
        for (int i = 0; i < 8; ++i) a[i]     = (__bf16)xrow[kbaseA + i];
        #pragma unroll
        for (int i = 0; i < 8; ++i) a[8 + i] = (__bf16)xrow[kbaseA + 16 + i];

        // ---- 16 B tiles (32x16 bf16) from LDS, one WMMA each.
        const int kbaseB = ks * 32 + half * 8;
        #pragma unroll
        for (int n = 0; n < 16; ++n) {
            const __bf16* wp = &Wt[(n * 16 + col) * LDSK + kbaseB];
            bf16x8 lo = *(const bf16x8*)(wp);
            bf16x8 hi = *(const bf16x8*)(wp + 16);
            bf16x16 bm;
            #pragma unroll
            for (int i = 0; i < 8; ++i) { bm[i] = lo[i]; bm[8 + i] = hi[i]; }
            acc[n] = __builtin_amdgcn_wmma_f32_16x16x32_bf16(
                false, a, false, bm, (short)0, acc[n], false, false);
        }
    }

    // ---- Fused epilogue: ps[r] = sum_n tanh(e[m][n] + b[n]) * uw[n],
    // with m = r + half*8 per the C-matrix VGPR layout.
    float ps[8];
    #pragma unroll
    for (int r = 0; r < 8; ++r) ps[r] = 0.0f;
    #pragma unroll
    for (int n = 0; n < 16; ++n) {
        const int e  = n * 16 + col;
        const float bv = b_lds[e];
        const float uv = uw_lds[e];
        #pragma unroll
        for (int r = 0; r < 8; ++r)
            ps[r] += tanhf(acc[n][r] + bv) * uv;
    }
    // Reduce over the 16 lanes of each half (wave32).
    #pragma unroll
    for (int r = 0; r < 8; ++r) {
        float v = ps[r];
        v += __shfl_xor(v, 1, 16);
        v += __shfl_xor(v, 2, 16);
        v += __shfl_xor(v, 4, 16);
        v += __shfl_xor(v, 8, 16);
        ps[r] = v;
    }
    if (col == 0) {
        #pragma unroll
        for (int r = 0; r < 8; ++r)
            scores[rowBase + half * 8 + r] = ps[r];
    }
}

// ---------------------------------------------------------------------------
// Kernel 2: per batch b: a = exp(s)*mask; a /= (sum_t a + eps);
//           out[b,d] = sum_t x[b,t,d] * a[t].  One block per batch.
// x is < 192 MB, so kernel 1's streaming read leaves it L2-resident for this
// second pass (default RT temporal hints on purpose).
// ---------------------------------------------------------------------------
__global__ __launch_bounds__(256) void attn_pool_kernel(
    const float* __restrict__ x, const float* __restrict__ scores,
    const int* __restrict__ mask, float* __restrict__ out)
{
    __shared__ float a_lds[TT];
    __shared__ float red[8];
    const int b   = blockIdx.x;
    const int tid = threadIdx.x;

    float psum = 0.0f;
    for (int t = tid; t < TT; t += 256) {
        float e = __expf(scores[b * TT + t]) * (float)mask[b * TT + t];
        a_lds[t] = e;
        psum += e;
    }
    psum += __shfl_xor(psum, 1);
    psum += __shfl_xor(psum, 2);
    psum += __shfl_xor(psum, 4);
    psum += __shfl_xor(psum, 8);
    psum += __shfl_xor(psum, 16);
    if ((tid & 31) == 0) red[tid >> 5] = psum;
    __syncthreads();
    float tot = 0.0f;
    #pragma unroll
    for (int w = 0; w < 8; ++w) tot += red[w];
    const float inv = 1.0f / (tot + 1e-7f);

    // Coalesced channel-wise weighted sum: thread tid owns channel d = tid.
    float acc = 0.0f;
    const float* __restrict__ xp = x + (long)b * TT * DD + tid;
    #pragma unroll 4
    for (int t = 0; t < TT; ++t)
        acc += xp[(long)t * DD] * a_lds[t];
    out[b * DD + tid] = acc * inv;
}

// ---------------------------------------------------------------------------
extern "C" void kernel_launch(void* const* d_in, const int* in_sizes, int n_in,
                              void* d_out, int out_size, void* d_ws, size_t ws_size,
                              hipStream_t stream) {
    const float* x    = (const float*)d_in[0];   // [B,T,D]
    const float* W    = (const float*)d_in[1];   // [D,D]
    const float* bvec = (const float*)d_in[2];   // [D]
    const float* uw   = (const float*)d_in[3];   // [D]
    const int*   mask = (const int*)d_in[4];     // [B,T]
    float* out    = (float*)d_out;               // [B,D]
    float* scores = (float*)d_ws;                // [B*T] fp32 = 512 KB scratch

    const int rows = BBATCH * TT;                // 131072
    const int blocks1 = rows / (8 * 16);         // 8 waves x 16 rows per block
    const size_t lds1 = (size_t)DD * LDSK * 2 + 2 * DD * sizeof(float);

    attn_scores_kernel<<<blocks1, 256, lds1, stream>>>(x, W, bvec, uw, scores);
    attn_pool_kernel<<<BBATCH, 256, 0, stream>>>(x, scores, mask, out);
}
